// TransformerSelfAttention_81209241632917
// MI455X (gfx1250) — compile-verified, matrix operand order checked by reading, not run on hardware
//
#include <hip/hip_runtime.h>
#include <hip/hip_bf16.h>

#define B_   2
#define S_   2048
#define HID_ 2048
#define NH_  16
#define HD_  128
#define ROT_ 64
#define MASKV (-10000.0f)

typedef __bf16 bf16_t;
typedef __attribute__((ext_vector_type(16))) __bf16 v16bf;
typedef __attribute__((ext_vector_type(8)))  __bf16 v8bf;
typedef __attribute__((ext_vector_type(4)))  __bf16 v4bf;
typedef __attribute__((ext_vector_type(8)))  float  v8f;
typedef __attribute__((ext_vector_type(4)))  int    v4i;
typedef v4i __attribute__((address_space(1)))* gptr_v4i;  // global int4*
typedef v4i __attribute__((address_space(3)))* lptr_v4i;  // LDS int4*

#if __has_builtin(__builtin_amdgcn_global_load_async_to_lds_b128) && \
    __has_builtin(__builtin_amdgcn_s_wait_asynccnt)
#define HAVE_ASYNC_LDS 1
#else
#define HAVE_ASYNC_LDS 0
#endif

// ---------------------------------------------------------------------------
// Fragment loader for 16-bit WMMA operands from a row-major (rows x K) source.
// A: lane = M row. B: lane = N column of KxN == row of the row-major NxK src.
// ISA layout => two contiguous 16-byte segments at row*ld + hi*8 and +16.
// ---------------------------------------------------------------------------
__device__ __forceinline__ v16bf load_frag16(const bf16_t* base, int ld) {
  const int lane = threadIdx.x & 31;
  const int row  = lane & 15;
  const int hi   = lane >> 4;
  const bf16_t* p = base + (size_t)row * (size_t)ld + hi * 8;
  v8bf lo = *(const v8bf*)(p);
  v8bf hv = *(const v8bf*)(p + 16);
  return __builtin_shufflevector(lo, hv, 0,1,2,3,4,5,6,7,8,9,10,11,12,13,14,15);
}

__device__ __forceinline__ v8f wmma_bf16(v16bf a, v16bf b, v8f c) {
  return __builtin_amdgcn_wmma_f32_16x16x32_bf16(
      false, a, false, b, (short)0, c, false, false);
}

// ---------------------------------------------------------------------------
// Kernel 1a: fp32 -> bf16 conversion, 4 elements/thread
// ---------------------------------------------------------------------------
__global__ void cvt_f32_bf16_kernel(const float* __restrict__ src,
                                    bf16_t* __restrict__ dst, int n4) {
  int i = blockIdx.x * blockDim.x + threadIdx.x;
  int stride = gridDim.x * blockDim.x;
  for (; i < n4; i += stride) {
    float4 f = ((const float4*)src)[i];
    v4bf o = { (bf16_t)f.x, (bf16_t)f.y, (bf16_t)f.z, (bf16_t)f.w };
    ((v4bf*)dst)[i] = o;
  }
}

// ---------------------------------------------------------------------------
// Kernel 1b: RoPE sin/cos table: tab[s*32 + f] = (sin, cos)(s * 10000^(-2f/64))
// ---------------------------------------------------------------------------
__global__ void rope_table_kernel(float2* __restrict__ tab) {
  int i = blockIdx.x * blockDim.x + threadIdx.x;  // i < S_*32
  if (i < S_ * 32) {
    int s = i >> 5, f = i & 31;
    float invf = __powf(10000.f, -((float)(2 * f)) * (1.0f / 64.f));
    float sn, cs;
    __sincosf((float)s * invf, &sn, &cs);
    tab[i] = make_float2(sn, cs);
  }
}

// ---------------------------------------------------------------------------
// Cooperative stage of a 32x32 bf16 A tile into LDS (2KB), 128 threads x 16B.
// Uses the CDNA5 async load-to-LDS path when available.
// ---------------------------------------------------------------------------
__device__ __forceinline__ void stage_A(const bf16_t* __restrict__ X, int m0,
                                        int k, bf16_t* lds) {
  const int t   = threadIdx.x;      // 0..127
  const int row = t >> 2;           // 0..31
  const int col = (t & 3) * 8;      // 0,8,16,24
  const bf16_t* g = X + (size_t)(m0 + row) * HID_ + k + col;
#if HAVE_ASYNC_LDS
  __builtin_amdgcn_global_load_async_to_lds_b128(
      (gptr_v4i)g, (lptr_v4i)(lds + t * 8), 0, 0);
#else
  *(v8bf*)(lds + t * 8) = *(const v8bf*)g;
#endif
}

// ---------------------------------------------------------------------------
// Kernel 2: fused QKV projection + bias + RoPE.
//   X (B*S,HID) bf16 ; W (3*HID,HID) bf16 ; outputs Q,K (B,NH,S,HD) bf16,
//   Vt (B,NH,HD,S) bf16.  Wave tile: 32x128; block = 4 waves = 32x512.
//   A tile double-buffered in LDS via async copies, shared by all 4 waves.
// ---------------------------------------------------------------------------
__global__ void __launch_bounds__(128)
qkv_rope_kernel(const bf16_t* __restrict__ X, const bf16_t* __restrict__ W,
                const float* __restrict__ bq, const float* __restrict__ bk,
                const float* __restrict__ bv, const float2* __restrict__ tab,
                bf16_t* __restrict__ Qb, bf16_t* __restrict__ Kb,
                bf16_t* __restrict__ VT) {
  const int wave  = threadIdx.x >> 5;
  const int lane  = threadIdx.x & 31;
  const int col16 = lane & 15;
  const int hi    = lane >> 4;
  const int m0 = blockIdx.x * 32;
  const int n0 = blockIdx.y * 512 + wave * 128;

  __shared__ __align__(16) bf16_t Atile[2][32 * 32];

  v8f acc[2][8] = {};
  stage_A(X, m0, 0, Atile[0]);
  for (int k = 0; k < HID_; k += 32) {
    const int cur = (k >> 5) & 1;
#if HAVE_ASYNC_LDS
    __builtin_amdgcn_s_wait_asynccnt(0);
#endif
    __syncthreads();
    if (k + 32 < HID_) stage_A(X, m0, k + 32, Atile[cur ^ 1]);
    v16bf a0 = load_frag16(Atile[cur], 32);
    v16bf a1 = load_frag16(Atile[cur] + 16 * 32, 32);
#pragma unroll
    for (int ni = 0; ni < 8; ++ni) {
      v16bf bf = load_frag16(W + (size_t)(n0 + ni * 16) * HID_ + k, HID_);
      acc[0][ni] = wmma_bf16(a0, bf, acc[0][ni]);
      acc[1][ni] = wmma_bf16(a1, bf, acc[1][ni]);
    }
  }

  const int which = n0 >> 11;  // 0=q, 1=k, 2=v (uniform per wave)
  const float* bias = (which == 0) ? bq : (which == 1) ? bk : bv;

#pragma unroll
  for (int ni = 0; ni < 8; ++ni) {
    const int n    = n0 + ni * 16 + col16;
    const int colm = n & (HID_ - 1);
    const int h    = colm >> 7;
    const int d    = colm & (HD_ - 1);
    const float bcol = bias[colm];
    const bool doRope = (which < 2) && (d < ROT_);  // uniform per wave
    const int f = d >> 1;
#pragma unroll
    for (int mi = 0; mi < 2; ++mi) {
#pragma unroll
      for (int r = 0; r < 8; ++r) {
        const int m = m0 + mi * 16 + r + 8 * hi;
        const int b = m >> 11;       // m / S
        const int s = m & (S_ - 1);  // m % S
        float val = acc[mi][ni][r] + bcol;
        float partner = __shfl_xor(val, 1, 32);  // neighbor head-dim d^1
        if (doRope) {
          const float2 scv = tab[s * 32 + f];  // (sin, cos)
          val = (d & 1) ? (val * scv.y + partner * scv.x)
                        : (val * scv.y - partner * scv.x);
        }
        const size_t head = (size_t)b * NH_ + h;
        if (which == 0)
          Qb[(head * S_ + s) * HD_ + d] = (bf16_t)val;
        else if (which == 1)
          Kb[(head * S_ + s) * HD_ + d] = (bf16_t)val;
        else
          VT[(head * HD_ + d) * S_ + s] = (bf16_t)val;
      }
    }
  }
}

// ---------------------------------------------------------------------------
// Kernel 3: causal flash attention, one wave per 16 query rows, 64-key blocks.
// ---------------------------------------------------------------------------
__global__ void __launch_bounds__(128)
flash_attn_kernel(const bf16_t* __restrict__ Qb, const bf16_t* __restrict__ Kb,
                  const bf16_t* __restrict__ VT, const float* __restrict__ amask,
                  float* __restrict__ out) {
  const int wave  = threadIdx.x >> 5;
  const int lane  = threadIdx.x & 31;
  const int col16 = lane & 15;
  const int hi    = lane >> 4;
  const int q0 = blockIdx.x * 64 + wave * 16;
  const int h  = blockIdx.y;
  const int b  = blockIdx.z;

  __shared__ __align__(16) bf16_t Pbuf[4][16 * 64];
  bf16_t* myP = Pbuf[wave];

  const size_t head = (size_t)b * NH_ + h;
  const bf16_t* Qbase = Qb + head * (size_t)S_ * HD_;
  const bf16_t* Kbase = Kb + head * (size_t)S_ * HD_;
  const bf16_t* Vbase = VT + head * (size_t)HD_ * S_;

  v16bf qf[4];
#pragma unroll
  for (int c = 0; c < 4; ++c)
    qf[c] = load_frag16(Qbase + (size_t)q0 * HD_ + c * 32, HD_);

  v8f acc[8] = {};
  float mrow[8], lrow[8];
#pragma unroll
  for (int r = 0; r < 8; ++r) { mrow[r] = -1e30f; lrow[r] = 0.f; }

  const float scale = 0.08838834764831845f;  // 1/sqrt(128)
  const int kend = (q0 + 16 + 63) & ~63;     // causal upper bound, 64-aligned

  for (int k0 = 0; k0 < kend; k0 += 64) {
    v8f sfr[4] = {};
#pragma unroll
    for (int c = 0; c < 4; ++c) {
#pragma unroll
      for (int t = 0; t < 4; ++t) {
        v16bf kf = load_frag16(Kbase + (size_t)(k0 + t * 16) * HD_ + c * 32, HD_);
        sfr[t] = wmma_bf16(qf[c], kf, sfr[t]);
      }
    }
    float amv[4];
#pragma unroll
    for (int t = 0; t < 4; ++t)
      amv[t] = amask[(size_t)b * S_ + k0 + t * 16 + col16];

    float sc[8];
#pragma unroll
    for (int r = 0; r < 8; ++r) {
      const int q = q0 + r + 8 * hi;
      float x[4];
#pragma unroll
      for (int t = 0; t < 4; ++t) {
        x[t] = sfr[t][r] * scale;
        if (k0 + t * 16 + col16 > q) x[t] = MASKV;
        x[t] += amv[t];
      }
      float mx = fmaxf(fmaxf(x[0], x[1]), fmaxf(x[2], x[3]));
#pragma unroll
      for (int o = 8; o >= 1; o >>= 1) mx = fmaxf(mx, __shfl_xor(mx, o, 32));
      const float mnew = fmaxf(mrow[r], mx);
      const float scl  = __expf(mrow[r] - mnew);
      mrow[r] = mnew;
      float rs = 0.f;
      const int row = r + 8 * hi;
#pragma unroll
      for (int t = 0; t < 4; ++t) {
        x[t] = __expf(x[t] - mnew);
        rs += x[t];
        myP[row * 64 + t * 16 + col16] = (bf16_t)x[t];  // P -> LDS (A layout src)
      }
#pragma unroll
      for (int o = 8; o >= 1; o >>= 1) rs += __shfl_xor(rs, o, 32);
      lrow[r] = lrow[r] * scl + rs;
      sc[r] = scl;
    }

    asm volatile("" ::: "memory");  // keep ds_store before ds_load
    v16bf pa0 = load_frag16(myP, 64);       // keys k0   .. k0+31
    v16bf pa1 = load_frag16(myP + 32, 64);  // keys k0+32.. k0+63

#pragma unroll
    for (int dt = 0; dt < 8; ++dt) {
#pragma unroll
      for (int r = 0; r < 8; ++r) acc[dt][r] *= sc[r];
      v16bf vf0 = load_frag16(Vbase + (size_t)(dt * 16) * S_ + k0, S_);
      v16bf vf1 = load_frag16(Vbase + (size_t)(dt * 16) * S_ + k0 + 32, S_);
      acc[dt] = wmma_bf16(pa0, vf0, acc[dt]);
      acc[dt] = wmma_bf16(pa1, vf1, acc[dt]);
    }
  }

#pragma unroll
  for (int dt = 0; dt < 8; ++dt) {
    const int d = dt * 16 + col16;
#pragma unroll
    for (int r = 0; r < 8; ++r) {
      const int q = q0 + r + 8 * hi;
      out[((size_t)b * S_ + q) * HID_ + h * HD_ + d] = acc[dt][r] / lrow[r];
    }
  }
}

// ---------------------------------------------------------------------------
extern "C" void kernel_launch(void* const* d_in, const int* in_sizes, int n_in,
                              void* d_out, int out_size, void* d_ws, size_t ws_size,
                              hipStream_t stream) {
  const float* hs = (const float*)d_in[0];
  const float* am = (const float*)d_in[1];
  const float* Wq = (const float*)d_in[2];
  const float* bq = (const float*)d_in[3];
  const float* Wk = (const float*)d_in[4];
  const float* bk = (const float*)d_in[5];
  const float* Wv = (const float*)d_in[6];
  const float* bv = (const float*)d_in[7];
  float* out = (float*)d_out;

  const size_t M  = (size_t)B_ * S_;      // 4096
  const size_t MK = M * HID_;             // 8388608
  const size_t WK = (size_t)HID_ * HID_;  // 4194304

  char* ws = (char*)d_ws;
  bf16_t* Xbf = (bf16_t*)ws;  ws += MK * 2;
  bf16_t* Wc  = (bf16_t*)ws;  ws += 3 * WK * 2;
  bf16_t* Qb  = (bf16_t*)ws;  ws += MK * 2;
  bf16_t* Kb  = (bf16_t*)ws;  ws += MK * 2;
  bf16_t* VT  = (bf16_t*)ws;  ws += MK * 2;
  float2* tab = (float2*)ws;  ws += (size_t)S_ * 32 * sizeof(float2);

  cvt_f32_bf16_kernel<<<2048, 256, 0, stream>>>(hs, Xbf, (int)(MK / 4));
  cvt_f32_bf16_kernel<<<1024, 256, 0, stream>>>(Wq, Wc,          (int)(WK / 4));
  cvt_f32_bf16_kernel<<<1024, 256, 0, stream>>>(Wk, Wc + WK,     (int)(WK / 4));
  cvt_f32_bf16_kernel<<<1024, 256, 0, stream>>>(Wv, Wc + 2 * WK, (int)(WK / 4));
  rope_table_kernel<<<(S_ * 32 + 255) / 256, 256, 0, stream>>>(tab);

  dim3 g2((unsigned)(M / 32), (unsigned)(3 * HID_ / 512));
  qkv_rope_kernel<<<g2, 128, 0, stream>>>(Xbf, Wc, bq, bk, bv, tab, Qb, Kb, VT);

  dim3 g3((unsigned)(S_ / 64), NH_, B_);
  flash_attn_kernel<<<g3, 128, 0, stream>>>(Qb, Kb, VT, am, out);
}
